// DynamicGATLayer_65833258713109
// MI455X (gfx1250) — matrix operand order repaired and checked by reading, not compiled, not run on hardware
//
#include <hip/hip_runtime.h>
#include <hip/hip_bf16.h>
#include <stddef.h>

#define BB  2
#define NN  2048
#define FF  128
#define FEE 8
#define HH  4
#define CC  64
#define HC  (HH*CC)

typedef __attribute__((ext_vector_type(16))) __bf16 v16bf;
typedef __attribute__((ext_vector_type(8)))  float  v8f;

static __device__ __forceinline__ __bf16 f2bf(float f) { return (__bf16)f; }

// ---------------------------------------------------------------------------
// Kernel 1: h = X * W_gat (bf16 WMMA), plus s_src/s_dst rank-1 terms.
// Block = 16 node rows, 4 waves = 4 heads. h stored bf16 [b][h][n][c] so the
// attention kernel's WMMA B-operand (lane%16 = c, elements = K rows) is, per element, a
// 32B load of 16 contiguous c across lanes.
// ---------------------------------------------------------------------------
__global__ __launch_bounds__(128) void gat_h_kernel(
    const float* __restrict__ X, const float* __restrict__ Wg,
    const float* __restrict__ asrc, const float* __restrict__ adst,
    __bf16* __restrict__ hB, float* __restrict__ ssrc, float* __restrict__ sdst)
{
    __shared__ float ldsH[HH][16 * CC];          // 16 KB
    const int tile = blockIdx.x;
    const int b    = tile / (NN / 16);
    const int i0   = (tile % (NN / 16)) * 16;
    const int hh   = threadIdx.x >> 5;           // head = wave
    const int lane = threadIdx.x & 31;
    const int m    = lane & 15;
    const int hi   = lane >> 4;
    const int kb   = hi * 8;                     // A-layout K base

    v8f acc[4] = {};
    for (int k0 = 0; k0 < FF; k0 += 32) {
        // A operand: X tile 16x32 (M=lane%16; K per documented 16-bit layout)
        const float* xr = X + ((size_t)(b * NN + i0 + m)) * FF + k0;
        v16bf a;
#pragma unroll
        for (int e = 0; e < 8; ++e) {
            a[e]     = f2bf(xr[kb + e]);
            a[8 + e] = f2bf(xr[16 + kb + e]);
        }
#pragma unroll
        for (int t = 0; t < 4; ++t) {
            // B operand: W_gat 32x16, lane%16 = column c, elements = K (A-layout order)
            const float* wr = Wg + ((size_t)hh * FF + k0) * CC + t * 16 + m;
            v16bf bv;
#pragma unroll
            for (int e = 0; e < 16; ++e) bv[e] = f2bf(wr[(size_t)((e < 8) ? (kb + e) : (16 + kb + (e - 8))) * CC]);
            acc[t] = __builtin_amdgcn_wmma_f32_16x16x32_bf16(
                false, a, false, bv, (short)0, acc[t], false, false);
        }
    }

    // D layout: element v -> row m = v + 8*hi, col n = lane%16
#pragma unroll
    for (int t = 0; t < 4; ++t)
#pragma unroll
        for (int v = 0; v < 8; ++v) {
            const int mo = v + 8 * hi;
            const int c  = t * 16 + (lane & 15);
            const float val = acc[t][v];
            ldsH[hh][mo * CC + c] = val;
            hB[((size_t)(b * HH + hh) * NN + i0 + mo) * CC + c] = f2bf(val);
        }

    // s_src / s_dst: row = lane%16, each half of the wave does 32 of 64 c's
    {
        const int row  = lane & 15;
        const int half = hi;
        float ss = 0.f, sd = 0.f;
#pragma unroll
        for (int c = 0; c < 32; ++c) {
            const int cc = half * 32 + c;
            const float hv = ldsH[hh][row * CC + cc];
            ss += hv * asrc[hh * CC + cc];
            sd += hv * adst[hh * CC + cc];
        }
        ss += __shfl_xor(ss, 16, 32);
        sd += __shfl_xor(sd, 16, 32);
        if (lane < 16) {
            ssrc[((size_t)(b * HH + hh)) * NN + i0 + row] = ss;
            sdst[((size_t)(b * HH + hh)) * NN + i0 + row] = sd;
        }
    }
}

// ---------------------------------------------------------------------------
// Kernel 2: fused edge-gate + mask + online softmax + attn*h (bf16 WMMA).
// Block = 16 rows (i-tile); 4 waves = 4 heads; loop over j in 32-wide tiles.
// E is read exactly once across the whole grid. No [B,N,N,H] intermediates.
// ---------------------------------------------------------------------------
__global__ __launch_bounds__(128) void gat_attn_kernel(
    const float* __restrict__ A, const float* __restrict__ E,
    const float* __restrict__ We, const float* __restrict__ be,
    const __bf16* __restrict__ hB, const float* __restrict__ ssrc,
    const float* __restrict__ sdstg, const float* __restrict__ bgat,
    float* __restrict__ out)
{
    __shared__ float sdstL[HH * NN];   // 32 KB: s_dst for all heads, this batch
    __shared__ float amask[16 * 32];   // 2 KB: -1e9/0 mask tile
    const int tile = blockIdx.x;
    const int b    = tile / (NN / 16);
    const int i0   = (tile % (NN / 16)) * 16;
    const int tid  = threadIdx.x;
    const int hh   = tid >> 5;
    const int lane = tid & 31;
    const int m    = lane & 15;
    const int hi   = lane >> 4;
    const int kb   = hi * 8;

    for (int idx = tid; idx < HH * NN; idx += 128)
        sdstL[idx] = sdstg[(size_t)b * HH * NN + idx];

    float we[8];
#pragma unroll
    for (int e = 0; e < 8; ++e) we[e] = We[e];
    const float bedge = be[0];

    const float my_ssrc = ssrc[((size_t)(b * HH + hh)) * NN + i0 + m];

    float rmax = -3.0e38f, rsum = 0.f;
    v8f acc[4] = {};

    __syncthreads();

    for (int j0 = 0; j0 < NN; j0 += 32) {
        // ---- cooperative: edge gate + adjacency mask for the 16x32 tile ----
#pragma unroll
        for (int r = 0; r < 4; ++r) {
            const int p  = r * 128 + tid;      // 512 (i,j) pairs / 128 threads
            const int mm = p >> 5;
            const int jj = p & 31;
            const float* er =
                E + (((size_t)b * NN + (i0 + mm)) * NN + (j0 + jj)) * FEE;
            float z = bedge;
#pragma unroll
            for (int e = 0; e < 8; ++e) z += er[e] * we[e];
            const float gate = 1.f / (1.f + __expf(-z));
            const float Ag =
                A[((size_t)b * NN + (i0 + mm)) * NN + (j0 + jj)] * gate;
            amask[mm * 32 + jj] = (Ag == 0.f) ? -1.0e9f : 0.f;
            if (j0 + 32 < NN)
                __builtin_prefetch(er + 32 * FEE, 0, 1);   // global_prefetch_b8
        }
        __syncthreads();

        // ---- per-head: logits + online softmax (flash-style) ----
        float l[16];
        float tmax = -3.0e38f;
#pragma unroll
        for (int e = 0; e < 16; ++e) {
            const int k = (e < 8) ? (kb + e) : (16 + kb + (e - 8)); // A layout
            float lg = my_ssrc + sdstL[hh * NN + j0 + k];
            lg = (lg >= 0.f) ? lg : 0.2f * lg;                 // leaky_relu
            lg += amask[m * 32 + k];
            l[e] = lg;
            tmax = fmaxf(tmax, lg);
        }
        tmax = fmaxf(tmax, __shfl_xor(tmax, 16, 32));          // full row max
        const float Mnew  = fmaxf(rmax, tmax);
        const float scale = __expf(rmax - Mnew);
        float tsum = 0.f;
        v16bf pa;
#pragma unroll
        for (int e = 0; e < 16; ++e) {
            const float pv = __expf(l[e] - Mnew);
            tsum += pv;
            pa[e] = f2bf(pv);
        }
        tsum += __shfl_xor(tsum, 16, 32);
        rsum  = rsum * scale + tsum;
        rmax  = Mnew;

        // rescale accumulators: element v belongs to row v+8*hi; row stats
        // live in lane (row%16) -> gather with shfl
#pragma unroll
        for (int v = 0; v < 8; ++v) {
            const float sc = __shfl(scale, v + 8 * hi, 32);
#pragma unroll
            for (int t = 0; t < 4; ++t) acc[t][v] *= sc;
        }

        // B operand: lane%16 = column c, elements = K rows j0+k (A-layout order)
        const __bf16* hrow = hB + ((size_t)(b * HH + hh) * NN + j0) * CC + m;
#pragma unroll
        for (int t = 0; t < 4; ++t) {
            v16bf bt;
#pragma unroll
            for (int e = 0; e < 16; ++e)
                bt[e] = hrow[(size_t)((e < 8) ? (kb + e) : (16 + kb + (e - 8))) * CC + t * 16];
            acc[t] = __builtin_amdgcn_wmma_f32_16x16x32_bf16(
                false, pa, false, bt, (short)0, acc[t], false, false);
        }
        __syncthreads();
    }

    // ---- epilogue: /rowsum, +bias, ELU ----
#pragma unroll
    for (int v = 0; v < 8; ++v) {
        const float rs  = __shfl(rsum, v + 8 * hi, 32);
        const float inv = 1.f / fmaxf(rs, 1e-20f);
        const int mo = v + 8 * hi;
        const int n  = lane & 15;
#pragma unroll
        for (int t = 0; t < 4; ++t) {
            float val = acc[t][v] * inv + bgat[hh * CC + t * 16 + n];
            val = (val > 0.f) ? val : (__expf(val) - 1.f);     // elu
            out[((size_t)(b * NN + i0 + mo)) * HC + hh * CC + t * 16 + n] = val;
        }
    }
}

// ---------------------------------------------------------------------------
// Kernel 3: LayerNorm over 256 features, in-place (block owns one row).
// ---------------------------------------------------------------------------
__global__ __launch_bounds__(256) void ln_kernel(
    float* __restrict__ out, const float* __restrict__ gamma,
    const float* __restrict__ beta)
{
    __shared__ float red[256];
    const int row = blockIdx.x;
    const int c   = threadIdx.x;
    const float x = out[(size_t)row * HC + c];

    red[c] = x;
    __syncthreads();
    for (int s = 128; s > 0; s >>= 1) {
        if (c < s) red[c] += red[c + s];
        __syncthreads();
    }
    const float mean = red[0] * (1.f / HC);
    __syncthreads();
    red[c] = x * x;
    __syncthreads();
    for (int s = 128; s > 0; s >>= 1) {
        if (c < s) red[c] += red[c + s];
        __syncthreads();
    }
    const float var = red[0] * (1.f / HC) - mean * mean;
    out[(size_t)row * HC + c] =
        (x - mean) * rsqrtf(var + 1e-3f) * gamma[c] + beta[c];
}

// ---------------------------------------------------------------------------
extern "C" void kernel_launch(void* const* d_in, const int* in_sizes, int n_in,
                              void* d_out, int out_size, void* d_ws, size_t ws_size,
                              hipStream_t stream)
{
    const float* X    = (const float*)d_in[0];
    const float* A    = (const float*)d_in[1];
    const float* E    = (const float*)d_in[2];
    const float* We   = (const float*)d_in[3];
    const float* be   = (const float*)d_in[4];
    const float* Wg   = (const float*)d_in[5];
    const float* asrc = (const float*)d_in[6];
    const float* adst = (const float*)d_in[7];
    const float* bgat = (const float*)d_in[8];
    const float* lng  = (const float*)d_in[9];
    const float* lnb  = (const float*)d_in[10];
    float* out = (float*)d_out;

    char* wsb = (char*)d_ws;
    __bf16* hB  = (__bf16*)wsb;                                  // 2 MB
    float*  ssr = (float*)(wsb + (size_t)BB * HH * NN * CC * 2); // 64 KB
    float*  sds = ssr + (size_t)BB * HH * NN;                    // 64 KB

    const dim3 grid1(BB * (NN / 16));
    gat_h_kernel<<<grid1, 128, 0, stream>>>(X, Wg, asrc, adst, hB, ssr, sds);
    gat_attn_kernel<<<grid1, 128, 0, stream>>>(A, E, We, be, hB, ssr, sds,
                                               bgat, out);
    ln_kernel<<<dim3(BB * NN), 256, 0, stream>>>(out, lng, lnb);
}
